// SpectralAIMoE_7464653161202
// MI455X (gfx1250) — matrix operand order repaired and checked by reading, not compiled, not run on hardware
//
#include <hip/hip_runtime.h>
#include <hip/hip_bf16.h>
#include <math.h>

typedef __attribute__((ext_vector_type(16))) __bf16 v16bf;
typedef __attribute__((ext_vector_type(8)))  __bf16 v8bf;
typedef __attribute__((ext_vector_type(8)))  float  v8f;
typedef __attribute__((ext_vector_type(4)))  float  v4f;

#define B_   4
#define S_   512
#define H_   512
#define I_   2048
#define E_   16
#define V_   32000
#define R_   256
#define T_   (B_*S_)

__device__ __forceinline__ __bf16 tobf(float f) { return (__bf16)f; }

__device__ __forceinline__ v8f wmma_bf16(v16bf a, v16bf b, v8f c) {
  return __builtin_amdgcn_wmma_f32_16x16x32_bf16(false, a, false, b, (short)0, c, false, false);
}

// ---- fragment loaders (row-major, K contiguous) ----
// A 16x32 bf16: lane<16 holds K 0..7 & 16..23; lane>=16 holds K 8..15 & 24..31
__device__ __forceinline__ v16bf frag_a_bf(const __bf16* __restrict__ p) {
  v8bf lo = *(const v8bf*)(p);
  v8bf hi = *(const v8bf*)(p + 16);
  return __builtin_shufflevector(lo, hi, 0,1,2,3,4,5,6,7,8,9,10,11,12,13,14,15);
}
// B 32x16 bf16: lane<16 holds K 0..7 & 16..23; lane>=16 holds K 8..15 & 24..31
__device__ __forceinline__ v16bf frag_b_bf(const __bf16* __restrict__ p) {
  v8bf lo = *(const v8bf*)(p);
  v8bf hi = *(const v8bf*)(p + 16);
  return __builtin_shufflevector(lo, hi, 0,1,2,3,4,5,6,7,8,9,10,11,12,13,14,15);
}

// ---------------- kernel 0: fp32 -> bf16 convert (emb) ----------------
__global__ void k_convert_bf16(const float* __restrict__ src, __bf16* __restrict__ dst, int n) {
  int i = (blockIdx.x * blockDim.x + threadIdx.x) * 4;
  if (i + 3 < n) {
    v4f v = *(const v4f*)(src + i);
    dst[i]     = tobf(v.x);
    dst[i + 1] = tobf(v.y);
    dst[i + 2] = tobf(v.z);
    dst[i + 3] = tobf(v.w);
  } else {
    for (int j = i; j < n; ++j) dst[j] = tobf(src[j]);
  }
}

// ---------------- kernel 0b: gather + convert selected expert weights ----------------
__global__ void k_gather_experts(const float* __restrict__ Wg, const float* __restrict__ Wu,
                                 const float* __restrict__ Wd, const int* __restrict__ tk_ids,
                                 __bf16* __restrict__ g_sel, __bf16* __restrict__ u_sel,
                                 __bf16* __restrict__ d_sel) {
  int pair  = blockIdx.y;
  int which = blockIdx.z;
  int e = tk_ids[pair];
  const float* src;
  __bf16* dst;
  size_t mat = (size_t)I_ * H_;
  if (which == 0)      { src = Wg + (size_t)e * mat; dst = g_sel + (size_t)pair * mat; }
  else if (which == 1) { src = Wu + (size_t)e * mat; dst = u_sel + (size_t)pair * mat; }
  else                 { src = Wd + (size_t)e * mat; dst = d_sel + (size_t)pair * mat; }
  int i = (blockIdx.x * blockDim.x + threadIdx.x) * 4;
  v4f v = *(const v4f*)(src + i);
  dst[i]     = tobf(v.x);
  dst[i + 1] = tobf(v.y);
  dst[i + 2] = tobf(v.z);
  dst[i + 3] = tobf(v.w);
}

// ---------------- kernel 1: embedding + pos + layernorm ----------------
__global__ void k_embed_ln(const int* __restrict__ ids, const float* __restrict__ emb,
                           const float* __restrict__ pos, const float* __restrict__ g,
                           const float* __restrict__ bta, float* __restrict__ hidden,
                           __bf16* __restrict__ hidden_bf) {
  int t = blockIdx.x;
  int s = t & (S_ - 1);
  int tid = threadIdx.x;
  int id = ids[t];
  const float* er = emb + (size_t)id * H_;
  const float* pr = pos + (size_t)s * H_;
  float v0 = er[tid]       + pr[tid];
  float v1 = er[tid + 256] + pr[tid + 256];
  __shared__ float r1[256], r2[256];
  r1[tid] = v0 + v1;
  r2[tid] = v0 * v0 + v1 * v1;
  __syncthreads();
  for (int w = 128; w > 0; w >>= 1) {
    if (tid < w) { r1[tid] += r1[tid + w]; r2[tid] += r2[tid + w]; }
    __syncthreads();
  }
  float mu  = r1[0] * (1.0f / H_);
  float var = r2[0] * (1.0f / H_) - mu * mu;
  float rs  = rsqrtf(var + 1e-5f);
  float y0 = (v0 - mu) * rs * g[tid]       + bta[tid];
  float y1 = (v1 - mu) * rs * g[tid + 256] + bta[tid + 256];
  size_t base = (size_t)t * H_;
  hidden[base + tid]          = y0;
  hidden[base + tid + 256]    = y1;
  hidden_bf[base + tid]       = tobf(y0);
  hidden_bf[base + tid + 256] = tobf(y1);
}

// ---------------- kernel 2: router + softmax + top-2 ----------------
__global__ void k_router(const float* __restrict__ hidden, const float* __restrict__ Wp,
                         const float* __restrict__ bp, const float* __restrict__ Wr,
                         float* __restrict__ probs_out, int* __restrict__ topk_ids,
                         float* __restrict__ topk_w) {
  int b = blockIdx.x, tid = threadIdx.x;
  __shared__ float prompt[H_];
  __shared__ float ri[R_];
  __shared__ float le[E_];
  const float* hb = hidden + (size_t)b * S_ * H_;
  float a0 = 0.f, a1 = 0.f;
  for (int s = 0; s < S_; ++s) {
    a0 += hb[(size_t)s * H_ + tid];
    a1 += hb[(size_t)s * H_ + tid + 256];
  }
  prompt[tid]       = a0 * (1.0f / S_);
  prompt[tid + 256] = a1 * (1.0f / S_);
  __syncthreads();
  float d = bp[tid];
  const float* wr = Wp + (size_t)tid * H_;
  for (int h = 0; h < H_; ++h) d += prompt[h] * wr[h];
  ri[tid] = d;
  __syncthreads();
  if (tid < E_) {
    float l = 0.f;
    const float* we = Wr + (size_t)tid * R_;
    for (int r = 0; r < R_; ++r) l += ri[r] * we[r];
    le[tid] = l;
  }
  __syncthreads();
  if (tid == 0) {
    float mx = le[0];
    for (int e = 1; e < E_; ++e) mx = fmaxf(mx, le[e]);
    float p[E_], sum = 0.f;
    for (int e = 0; e < E_; ++e) { p[e] = __expf(le[e] - mx); sum += p[e]; }
    float inv = 1.0f / sum;
    int i0 = 0;
    for (int e = 1; e < E_; ++e) if (p[e] > p[i0]) i0 = e;
    int i1 = (i0 == 0) ? 1 : 0;
    for (int e = 0; e < E_; ++e) if (e != i0 && p[e] > p[i1]) i1 = e;
    for (int e = 0; e < E_; ++e) probs_out[b * E_ + e] = p[e] * inv;
    float p0 = p[i0] * inv, p1 = p[i1] * inv;
    float rn = 1.0f / (p0 + p1 + 1e-8f);
    topk_ids[b * 2]     = i0;
    topk_ids[b * 2 + 1] = i1;
    topk_w[b * 2]       = p0 * rn;
    topk_w[b * 2 + 1]   = p1 * rn;
  }
}

// ---------------- kernel 3: gate/up GEMM + weighted SwiGLU epilogue ----------------
// 256 thr = 8 waves (2x4). Wave tile 32x32 dual acc. Block tile 64x128. Double-buffered.
// ~160 VGPRs: accG/accU 64 + frags 96 -> full pipelining within 256-VGPR window.
__global__ void __launch_bounds__(256)
k_gateup(const __bf16* __restrict__ hidden_bf, const __bf16* __restrict__ g_sel,
         const __bf16* __restrict__ u_sel, const float* __restrict__ topk_w,
         __bf16* __restrict__ act_bf) {
  int pair = blockIdx.z;
  int b = pair >> 1;
  float w = topk_w[pair];
  const __bf16* sa = hidden_bf + (size_t)b * S_ * H_;
  const __bf16* sg = g_sel + (size_t)pair * I_ * H_;
  const __bf16* su = u_sel + (size_t)pair * I_ * H_;
  __bf16* Cp = act_bf + (size_t)pair * S_ * I_;

  int lane = threadIdx.x & 31, wid = threadIdx.x >> 5;
  int lhalf = lane >> 4, lmod = lane & 15;
  int wm = wid & 1, wn = wid >> 1;
  int m0 = blockIdx.y * 64  + wm * 32;
  int n0 = blockIdx.x * 128 + wn * 32;

  unsigned aoff = (unsigned)(m0 + lmod) * H_ + lhalf * 8;
  unsigned boff = (unsigned)(n0 + lmod) * H_ + lhalf * 8;

  v8f accG[2][2] = {};
  v8f accU[2][2] = {};
  v16bf a0[2], g0[2], u0[2], a1[2], g1[2], u1[2];
#pragma unroll
  for (int i = 0; i < 2; ++i) a0[i] = frag_a_bf(sa + aoff + i * (16 * H_));
#pragma unroll
  for (int j = 0; j < 2; ++j) {
    g0[j] = frag_b_bf(sg + boff + j * (16 * H_));
    u0[j] = frag_b_bf(su + boff + j * (16 * H_));
  }
#pragma unroll 1
  for (int k0 = 0; k0 < H_; k0 += 64) {
#pragma unroll
    for (int i = 0; i < 2; ++i) a1[i] = frag_a_bf(sa + aoff + 32 + i * (16 * H_));
#pragma unroll
    for (int j = 0; j < 2; ++j) {
      g1[j] = frag_b_bf(sg + boff + 32 + j * (16 * H_));
      u1[j] = frag_b_bf(su + boff + 32 + j * (16 * H_));
    }
#pragma unroll
    for (int j = 0; j < 2; ++j) {
      accG[0][j] = wmma_bf16(a0[0], g0[j], accG[0][j]);
      accG[1][j] = wmma_bf16(a0[1], g0[j], accG[1][j]);
      accU[0][j] = wmma_bf16(a0[0], u0[j], accU[0][j]);
      accU[1][j] = wmma_bf16(a0[1], u0[j], accU[1][j]);
    }
    sa += 64; sg += 64; su += 64;
    if (k0 + 64 < H_) {
#pragma unroll
      for (int i = 0; i < 2; ++i) a0[i] = frag_a_bf(sa + aoff + i * (16 * H_));
#pragma unroll
      for (int j = 0; j < 2; ++j) {
        g0[j] = frag_b_bf(sg + boff + j * (16 * H_));
        u0[j] = frag_b_bf(su + boff + j * (16 * H_));
      }
    }
#pragma unroll
    for (int j = 0; j < 2; ++j) {
      accG[0][j] = wmma_bf16(a1[0], g1[j], accG[0][j]);
      accG[1][j] = wmma_bf16(a1[1], g1[j], accG[1][j]);
      accU[0][j] = wmma_bf16(a1[0], u1[j], accU[0][j]);
      accU[1][j] = wmma_bf16(a1[1], u1[j], accU[1][j]);
    }
  }
#pragma unroll
  for (int i = 0; i < 2; ++i)
#pragma unroll
    for (int j = 0; j < 2; ++j)
#pragma unroll
      for (int r = 0; r < 8; ++r) {
        int row = m0 + 16 * i + r + lhalf * 8;
        int col = n0 + 16 * j + lmod;
        float gv = accG[i][j][r];
        float uv = accU[i][j][r];
        float sg_ = gv / (1.0f + __expf(-gv));   // silu
        Cp[(size_t)row * I_ + col] = tobf(w * sg_ * uv);
      }
}

// ---------------- kernel 4: down GEMM per (b,k) pair -> partial eout ----------------
// 256 thr = 8 waves (2x4). Wave tile 64x32. Block tile 128x128. Double-buffered.
__global__ void __launch_bounds__(256)
k_down(const __bf16* __restrict__ act_bf, const __bf16* __restrict__ d_sel,
       float* __restrict__ eout_part) {
  int pair = blockIdx.z;
  int lane = threadIdx.x & 31, wid = threadIdx.x >> 5;
  int lhalf = lane >> 4, lmod = lane & 15;
  int wm = wid & 1, wn = wid >> 1;
  int m0 = blockIdx.y * 128 + wm * 64;
  int n0 = blockIdx.x * 128 + wn * 32;

  const __bf16* sa = act_bf + (size_t)pair * S_ * I_;   // act already scaled by topk_w
  const __bf16* sb = d_sel  + (size_t)pair * H_ * I_;
  unsigned aoff = (unsigned)(m0 + lmod) * I_ + lhalf * 8;
  unsigned boff = (unsigned)(n0 + lmod) * I_ + lhalf * 8;

  v8f acc[4][2] = {};
  v16bf a0[4], b0[2], a1[4], b1[2];
#pragma unroll
  for (int i = 0; i < 4; ++i) a0[i] = frag_a_bf(sa + aoff + i * (16 * I_));
#pragma unroll
  for (int j = 0; j < 2; ++j) b0[j] = frag_b_bf(sb + boff + j * (16 * I_));
#pragma unroll 1
  for (int k0 = 0; k0 < I_; k0 += 64) {
#pragma unroll
    for (int i = 0; i < 4; ++i) a1[i] = frag_a_bf(sa + aoff + 32 + i * (16 * I_));
#pragma unroll
    for (int j = 0; j < 2; ++j) b1[j] = frag_b_bf(sb + boff + 32 + j * (16 * I_));
#pragma unroll
    for (int i = 0; i < 4; ++i)
#pragma unroll
      for (int j = 0; j < 2; ++j) acc[i][j] = wmma_bf16(a0[i], b0[j], acc[i][j]);
    sa += 64; sb += 64;
    if (k0 + 64 < I_) {
#pragma unroll
      for (int i = 0; i < 4; ++i) a0[i] = frag_a_bf(sa + aoff + i * (16 * I_));
#pragma unroll
      for (int j = 0; j < 2; ++j) b0[j] = frag_b_bf(sb + boff + j * (16 * I_));
    }
#pragma unroll
    for (int i = 0; i < 4; ++i)
#pragma unroll
      for (int j = 0; j < 2; ++j) acc[i][j] = wmma_bf16(a1[i], b1[j], acc[i][j]);
  }
#pragma unroll
  for (int i = 0; i < 4; ++i)
#pragma unroll
    for (int j = 0; j < 2; ++j)
#pragma unroll
      for (int r = 0; r < 8; ++r) {
        int row = m0 + 16 * i + r + lhalf * 8;
        int col = n0 + 16 * j + lmod;
        eout_part[((size_t)pair * S_ + row) * H_ + col] = acc[i][j][r];
      }
}

// ---------------- kernel 5: combine pair partials + alpha blend -> hp (bf16) ----------------
__global__ void k_blend(const float* __restrict__ hidden, const float* __restrict__ eout_part,
                        const float* __restrict__ bgw, const float* __restrict__ bgb,
                        __bf16* __restrict__ hp_bf) {
  int t = blockIdx.x, tid = threadIdx.x;
  int b = t >> 9;          // t / S_
  int s = t & (S_ - 1);
  __shared__ float red[256];
  const float* h  = hidden + (size_t)t * H_;
  const float* c0 = eout_part + ((size_t)(b * 2)     * S_ + s) * H_;
  const float* c1 = eout_part + ((size_t)(b * 2 + 1) * S_ + s) * H_;
  red[tid] = h[tid] * bgw[tid] + h[tid + 256] * bgw[tid + 256];
  __syncthreads();
  for (int w = 128; w > 0; w >>= 1) {
    if (tid < w) red[tid] += red[tid + w];
    __syncthreads();
  }
  float alpha = 1.0f / (1.0f + __expf(-(red[0] + bgb[0])));
  float hp0 = (1.0f - alpha) * h[tid]       + alpha * (c0[tid]       + c1[tid]);
  float hp1 = (1.0f - alpha) * h[tid + 256] + alpha * (c0[tid + 256] + c1[tid + 256]);
  hp_bf[(size_t)t * H_ + tid]       = tobf(hp0);
  hp_bf[(size_t)t * H_ + tid + 256] = tobf(hp1);
}

// ---------------- kernel 6: fused logits GEMM hp @ emb^T ----------------
// 256 thr = 8 waves (2x4). Wave tile 64x32. Block tile 128x128. Double-buffered.
__global__ void __launch_bounds__(256)
k_logits(const __bf16* __restrict__ hp_bf, const __bf16* __restrict__ emb_bf,
         float* __restrict__ out) {
  int lane = threadIdx.x & 31, wid = threadIdx.x >> 5;
  int lhalf = lane >> 4, lmod = lane & 15;
  int wm = wid & 1, wn = wid >> 1;
  int m0 = blockIdx.y * 128 + wm * 64;
  int n0 = blockIdx.x * 128 + wn * 32;

  const __bf16* sa = hp_bf;
  const __bf16* sb = emb_bf;
  unsigned aoff = (unsigned)(m0 + lmod) * H_ + lhalf * 8;
  unsigned boff = (unsigned)(n0 + lmod) * H_ + lhalf * 8;

  v8f acc[4][2] = {};
  v16bf a0[4], b0[2], a1[4], b1[2];
#pragma unroll
  for (int i = 0; i < 4; ++i) a0[i] = frag_a_bf(sa + aoff + i * (16 * H_));
#pragma unroll
  for (int j = 0; j < 2; ++j) b0[j] = frag_b_bf(sb + boff + j * (16 * H_));
#pragma unroll 1
  for (int k0 = 0; k0 < H_; k0 += 64) {
#pragma unroll
    for (int i = 0; i < 4; ++i) a1[i] = frag_a_bf(sa + aoff + 32 + i * (16 * H_));
#pragma unroll
    for (int j = 0; j < 2; ++j) b1[j] = frag_b_bf(sb + boff + 32 + j * (16 * H_));
#pragma unroll
    for (int i = 0; i < 4; ++i)
#pragma unroll
      for (int j = 0; j < 2; ++j) acc[i][j] = wmma_bf16(a0[i], b0[j], acc[i][j]);
    sa += 64; sb += 64;
    if (k0 + 64 < H_) {
#pragma unroll
      for (int i = 0; i < 4; ++i) a0[i] = frag_a_bf(sa + aoff + i * (16 * H_));
#pragma unroll
      for (int j = 0; j < 2; ++j) b0[j] = frag_b_bf(sb + boff + j * (16 * H_));
    }
#pragma unroll
    for (int i = 0; i < 4; ++i)
#pragma unroll
      for (int j = 0; j < 2; ++j) acc[i][j] = wmma_bf16(a1[i], b1[j], acc[i][j]);
  }
#pragma unroll
  for (int i = 0; i < 4; ++i)
#pragma unroll
    for (int j = 0; j < 2; ++j)
#pragma unroll
      for (int r = 0; r < 8; ++r) {
        int row = m0 + 16 * i + r + lhalf * 8;
        int col = n0 + 16 * j + lmod;
        out[(size_t)row * V_ + col] = acc[i][j][r];
      }
}

extern "C" void kernel_launch(void* const* d_in, const int* in_sizes, int n_in,
                              void* d_out, int out_size, void* d_ws, size_t ws_size,
                              hipStream_t stream) {
  const int*   ids  = (const int*)  d_in[0];
  const float* emb  = (const float*)d_in[1];
  const float* pos  = (const float*)d_in[2];
  const float* ln_g = (const float*)d_in[3];
  const float* ln_b = (const float*)d_in[4];
  const float* Wp   = (const float*)d_in[5];
  const float* bp   = (const float*)d_in[6];
  const float* Wr   = (const float*)d_in[7];
  const float* Wg   = (const float*)d_in[8];
  const float* Wu   = (const float*)d_in[9];
  const float* Wd   = (const float*)d_in[10];
  const float* bgw  = (const float*)d_in[11];
  const float* bgb  = (const float*)d_in[12];
  float* out = (float*)d_out;

  char* ws = (char*)d_ws;
  size_t o = 0;
  float*  hidden    = (float*) (ws + o); o += (size_t)T_ * H_ * 4;
  __bf16* hidden_bf = (__bf16*)(ws + o); o += (size_t)T_ * H_ * 2;
  __bf16* emb_bf    = (__bf16*)(ws + o); o += (size_t)V_ * H_ * 2;
  __bf16* act_bf    = (__bf16*)(ws + o); o += (size_t)8 * S_ * I_ * 2;
  float*  eout_part = (float*) (ws + o); o += (size_t)8 * S_ * H_ * 4;
  __bf16* hp_bf     = (__bf16*)(ws + o); o += (size_t)T_ * H_ * 2;
  __bf16* g_sel     = (__bf16*)(ws + o); o += (size_t)8 * I_ * H_ * 2;
  __bf16* u_sel     = (__bf16*)(ws + o); o += (size_t)8 * I_ * H_ * 2;
  __bf16* d_sel     = (__bf16*)(ws + o); o += (size_t)8 * H_ * I_ * 2;
  int*    tk_ids    = (int*)   (ws + o); o += 64;
  float*  tk_w      = (float*) (ws + o); o += 64;

  k_convert_bf16<<<(V_ * H_) / 1024, 256, 0, stream>>>(emb, emb_bf, V_ * H_);
  k_embed_ln<<<T_, 256, 0, stream>>>(ids, emb, pos, ln_g, ln_b, hidden, hidden_bf);
  k_router<<<B_, 256, 0, stream>>>(hidden, Wp, bp, Wr, out + (size_t)T_ * V_, tk_ids, tk_w);
  k_gather_experts<<<dim3((I_ * H_) / 1024, 8, 3), 256, 0, stream>>>(Wg, Wu, Wd, tk_ids,
                                                                     g_sel, u_sel, d_sel);
  k_gateup<<<dim3(I_ / 128, S_ / 64, 8), 256, 0, stream>>>(hidden_bf, g_sel, u_sel, tk_w, act_bf);
  k_down<<<dim3(H_ / 128, S_ / 128, 8), 256, 0, stream>>>(act_bf, d_sel, eout_part);
  k_blend<<<T_, 256, 0, stream>>>(hidden, eout_part, bgw, bgb, hp_bf);
  k_logits<<<dim3(V_ / 128, T_ / 128), 256, 0, stream>>>(hp_bf, emb_bf, out);
}
